// BiLSTM_CRF_63788854280910
// MI455X (gfx1250) — matrix operand order repaired and checked by reading, not compiled, not run on hardware
//
#include <hip/hip_runtime.h>

// ---------------------------------------------------------------------------
// CRF loss for BiLSTM-CRF, CDNA5 (gfx1250, wave32, WMMA).
// Forward recursion as rescaled f16 GEMM against constant E = exp(transitions):
//   state: alpha[b,j] = M_b*ln2 + ln(P[b,j]),  M_b int exponent, P mantissa
//   step : S = P @ E^T (WMMA),  Q = S * exp(feat_t),
//          e = f16-exponent(rowmax Q),  P' = Q * 2^-e,  M' = M + e  (masked)
// Row-max via packed-f16 butterfly (raw ds_bpermute, hoisted addresses).
// t-loop unrolled x2 to ping-pong the pipelined exp(feat) buffers.
// One wave per 16-row batch tile; E fragments resident in VGPRs.
// ---------------------------------------------------------------------------

typedef __attribute__((ext_vector_type(16))) _Float16 v16h;
typedef __attribute__((ext_vector_type(8)))  _Float16 v8h;
typedef __attribute__((ext_vector_type(2)))  __fp16   v2h;   // matches cvt_pkrtz
typedef __attribute__((ext_vector_type(8)))  float    v8f;

#define CRF_T     512
#define CRF_B     1024
#define CRF_K     48
#define CRF_START 46
#define CRF_STOP  47
#define CRF_NEG   (-10000.0f)
#define LOG2E     1.44269504088896340736f
#define LN2       0.69314718055994530942f

#define NTILES 3            // 48 columns / 16
#define WAVES  4            // waves (batch tiles) per block
#define BTILES (CRF_B / 16) // 64 batch tiles
#define LDS_STRIDE 80       // f16 per LDS row: 160B -> 16B-aligned b128 loads

union V16U { v16h v; _Float16 h[16]; };
union V8FU { v8f  v; float    f[8];  };
union V2HU { v2h  v; int      i;     };

__device__ __forceinline__ float bperm_f(int addr, float x) {
    return __int_as_float(__builtin_amdgcn_ds_bpermute(addr, __float_as_int(x)));
}
__device__ __forceinline__ v2h pkmax(v2h a, v2h b) {
    return __builtin_elementwise_max(a, b);
}

__global__ __launch_bounds__(WAVES * 32)
void crf_forward_kernel(const float* __restrict__ feats,
                        const float* __restrict__ trans,
                        const int*   __restrict__ lengths,
                        float*       __restrict__ tile_logz)
{
    __shared__ _Float16 lds_ea[WAVES][16][LDS_STRIDE];

    const int lane = threadIdx.x & 31;
    const int wave = threadIdx.x >> 5;
    const int tile = blockIdx.x * WAVES + wave;
    const int base = tile * 16;          // first batch row of this tile
    const int half = lane >> 4;          // 0: lanes 0-15, 1: lanes 16-31
    const int col  = lane & 15;

    // Hoisted xor-butterfly bpermute addresses (bytes).
    int xaddr[4];
#pragma unroll
    for (int k = 0; k < 4; ++k) xaddr[k] = (lane ^ (1 << k)) << 2;
    const int xaddr16 = (lane ^ 16) << 2;

    // ---- Constant B fragments: E^T padded to 64x48, f16.
    // B layout (32x16 f16): N = lane&15 (=> j), K = (lane>>4)*16 + idx (=> i).
    v16h Bfrag[NTILES][2];
#pragma unroll
    for (int nt = 0; nt < NTILES; ++nt) {
#pragma unroll
        for (int kt = 0; kt < 2; ++kt) {
            V16U u;
#pragma unroll
            for (int idx = 0; idx < 16; ++idx) {
                const int i = kt * 32 + half * 16 + idx;   // prev-tag dim
                const int j = nt * 16 + col;               // next-tag dim
                const float tv = (i < CRF_K) ? trans[j * CRF_K + i] : CRF_NEG;
                u.h[idx] = (_Float16)__expf(tv);           // exp(-1e4) == 0 pad
            }
            Bfrag[nt][kt] = u.v;
        }
    }

    // Per-lane row metadata (C/D layout: VGPR v holds row M = half*8 + v).
    int lenr[8];
#pragma unroll
    for (int v = 0; v < 8; ++v) lenr[v] = lengths[base + half * 8 + v];

    float wstop[NTILES];   // exp(trans[STOP, j]) for the final logsumexp
#pragma unroll
    for (int nt = 0; nt < NTILES; ++nt)
        wstop[nt] = __expf(trans[CRF_STOP * CRF_K + nt * 16 + col]);

    // State: P[nt][v] mantissa for (b = base+half*8+v, j = nt*16+col); Mex[v] exponent.
    float P[NTILES][8];
    int   Mex[8];
#pragma unroll
    for (int nt = 0; nt < NTILES; ++nt) {
        const float p0 = ((nt * 16 + col) == CRF_START) ? 1.0f : 0.0f;
#pragma unroll
        for (int v = 0; v < 8; ++v) P[nt][v] = p0;
    }
#pragma unroll
    for (int v = 0; v < 8; ++v) Mex[v] = 0;

    // Zero this wave's LDS tile once (cols 48..63 + pad stay zero forever).
    {
        unsigned* z = (unsigned*)&lds_ea[wave][0][0];
        for (int x = lane; x < 16 * LDS_STRIDE / 2; x += 32) z[x] = 0u;
    }

    // Pipelined feat factor w = exp(feat) = exp2(feat * log2e); preload t = 0.
    float wA[NTILES][8], wB[NTILES][8];
    {
        const float* fb = feats + (size_t)base * CRF_K;
#pragma unroll
        for (int nt = 0; nt < NTILES; ++nt)
#pragma unroll
            for (int v = 0; v < 8; ++v)
                wA[nt][v] = __builtin_amdgcn_exp2f(
                    LOG2E * fb[(half * 8 + v) * CRF_K + nt * 16 + col]);
    }

    const v8f czero = {};

    // One forward step: uses wcur = exp(feat_t), fills wnext = exp(feat_{t+1}).
    auto step = [&](int t, float (&wcur)[NTILES][8], float (&wnext)[NTILES][8]) {
        // ---- scatter current mantissas P -> LDS (f16, A-major tile)
#pragma unroll
        for (int nt = 0; nt < NTILES; ++nt)
#pragma unroll
            for (int v = 0; v < 8; ++v)
                lds_ea[wave][half * 8 + v][nt * 16 + col] = (_Float16)P[nt][v];

        // ---- gather A fragments (16x32 f16 each; row = col, K = half*16 + idx, same order as B)
        const _Float16* rowp = &lds_ea[wave][col][0];
        const v8h a00 = *(const v8h*)(rowp +  0 + half * 16);
        const v8h a01 = *(const v8h*)(rowp +  8 + half * 16);
        const v8h a10 = *(const v8h*)(rowp + 32 + half * 16);
        const v8h a11 = *(const v8h*)(rowp + 40 + half * 16);
        const v16h A0 = __builtin_shufflevector(a00, a01, 0,1,2,3,4,5,6,7,8,9,10,11,12,13,14,15);
        const v16h A1 = __builtin_shufflevector(a10, a11, 0,1,2,3,4,5,6,7,8,9,10,11,12,13,14,15);

        // ---- software-pipeline next step's feat factors (unconditional loads)
        const int tn = (t + 1 < CRF_T) ? t + 1 : t;
        const float* fbn = feats + ((size_t)tn * CRF_B + base) * CRF_K;
#pragma unroll
        for (int nt = 0; nt < NTILES; ++nt)
#pragma unroll
            for (int v = 0; v < 8; ++v)
                wnext[nt][v] = __builtin_amdgcn_exp2f(
                    LOG2E * fbn[(half * 8 + v) * CRF_K + nt * 16 + col]);

        // ---- S = P @ E^T (6 WMMAs), then Q = S * exp(feat_t)
        float Q[NTILES][8];
#pragma unroll
        for (int nt = 0; nt < NTILES; ++nt) {
            v8f acc = __builtin_amdgcn_wmma_f32_16x16x32_f16(
                false, A0, false, Bfrag[nt][0], (short)0, czero, false, false);
            acc = __builtin_amdgcn_wmma_f32_16x16x32_f16(
                false, A1, false, Bfrag[nt][1], (short)0, acc, false, false);
            V8FU cu; cu.v = acc;
#pragma unroll
            for (int v = 0; v < 8; ++v) Q[nt][v] = cu.f[v] * wcur[nt][v];
        }

        // ---- packed-f16 row-max butterfly (rows 2p, 2p+1 share a register)
        int r[4];
#pragma unroll
        for (int p = 0; p < 4; ++p) {
            const float m0 = fmaxf(fmaxf(Q[0][2*p],   Q[1][2*p]),   Q[2][2*p]);
            const float m1 = fmaxf(fmaxf(Q[0][2*p+1], Q[1][2*p+1]), Q[2][2*p+1]);
            V2HU u; u.v = __builtin_amdgcn_cvt_pkrtz(m0, m1);
            r[p] = u.i;
        }
#pragma unroll
        for (int k = 0; k < 4; ++k) {
#pragma unroll
            for (int p = 0; p < 4; ++p) {
                V2HU a, b;
                a.i = r[p];
                b.i = __builtin_amdgcn_ds_bpermute(xaddr[k], r[p]);
                a.v = pkmax(a.v, b.v);
                r[p] = a.i;
            }
        }

        // ---- exponent-only renormalization + length mask (branchless)
#pragma unroll
        for (int v = 0; v < 8; ++v) {
            const unsigned bits = (unsigned)r[v >> 1];
            const unsigned ef   = (bits >> (10 + 16 * (v & 1))) & 0x1Fu; // f16 exp
            const int   e       = (int)ef - 14;                          // frexp exp
            const float scale   = __uint_as_float((141u - ef) << 23);    // 2^-e
            const bool  sel     = t < lenr[v];
#pragma unroll
            for (int nt = 0; nt < NTILES; ++nt) {
                const float Pn = Q[nt][v] * scale;
                P[nt][v] = sel ? Pn : P[nt][v];
            }
            Mex[v] = sel ? Mex[v] + e : Mex[v];
        }
    };

    for (int t = 0; t < CRF_T; t += 2) {
        step(t,     wA, wB);
        step(t + 1, wB, wA);
    }

    // ---- log_z per row: ln(sum_j P_j * 2^M * exp(ts_j)) = M*ln2 + ln(sum P*wstop)
    float lzsum = 0.0f;
#pragma unroll
    for (int v = 0; v < 8; ++v) {
        float z = P[0][v] * wstop[0] + P[1][v] * wstop[1] + P[2][v] * wstop[2];
#pragma unroll
        for (int k = 0; k < 4; ++k) z += bperm_f(xaddr[k], z);
        lzsum += (float)Mex[v] * LN2 + __logf(z);
    }
    // low half holds rows 0-7, high half rows 8-15; combine and store.
    const float total = lzsum + bperm_f(xaddr16, lzsum);
    if (lane == 0) tile_logz[tile] = total;
}

// ---------------------------------------------------------------------------
// Gold path score: one thread per batch element, deterministic block reduce.
// ---------------------------------------------------------------------------
__global__ __launch_bounds__(256)
void crf_gold_kernel(const float* __restrict__ feats,
                     const float* __restrict__ trans,
                     const int*   __restrict__ tags,
                     const int*   __restrict__ lengths,
                     float*       __restrict__ gold_part)
{
    __shared__ float red[256];
    const int tid = threadIdx.x;
    const int b   = blockIdx.x * 256 + tid;

    const int len = lengths[b];
    int prev = CRF_START;
    float g = 0.0f;
    for (int t = 0; t < len; ++t) {
        const int nxt = tags[b * CRF_T + t];
        g += trans[nxt * CRF_K + prev] +
             feats[((size_t)t * CRF_B + b) * CRF_K + nxt];
        prev = nxt;
    }
    g += trans[CRF_STOP * CRF_K + prev];

    red[tid] = g;
    __syncthreads();
    for (int s = 128; s > 0; s >>= 1) {
        if (tid < s) red[tid] += red[tid + s];
        __syncthreads();
    }
    if (tid == 0) gold_part[blockIdx.x] = red[0];
}

__global__ void crf_final_kernel(const float* __restrict__ ws,
                                 float* __restrict__ out)
{
    float s = 0.0f;
    for (int i = 0; i < BTILES; ++i) s += ws[i];
    for (int i = 0; i < CRF_B / 256; ++i) s -= ws[BTILES + i];
    out[0] = s;
}

extern "C" void kernel_launch(void* const* d_in, const int* in_sizes, int n_in,
                              void* d_out, int out_size, void* d_ws, size_t ws_size,
                              hipStream_t stream) {
    const float* feats   = (const float*)d_in[0];   // (T, B, K) f32
    const float* trans   = (const float*)d_in[1];   // (K, K)    f32
    const int*   tags    = (const int*)  d_in[2];   // (B, T)    i32
    const int*   lengths = (const int*)  d_in[3];   // (B,)      i32
    float* out = (float*)d_out;
    float* ws  = (float*)d_ws;                      // [0..63] logz, [64..67] gold

    crf_forward_kernel<<<BTILES / WAVES, WAVES * 32, 0, stream>>>(feats, trans, lengths, ws);
    crf_gold_kernel<<<CRF_B / 256, 256, 0, stream>>>(feats, trans, tags, lengths, ws + BTILES);
    crf_final_kernel<<<1, 1, 0, stream>>>(ws, out);
    (void)in_sizes; (void)n_in; (void)out_size; (void)ws_size;
}
